// POLLADecoderLayer_85899345920254
// MI455X (gfx1250) — compile-verified, matrix operand order checked by reading, not run on hardware
//
#include <hip/hip_runtime.h>

// ---------------------------------------------------------------------------
// CDNA5 (gfx1250) bf16 WMMA decoder layer.
// D = A*B + C via v_wmma_f32_16x16x32_bf16, wave32, 32x32 per-wave tiles,
// float4 global staging + packed bf16 LDS (32-bit pair accesses throughout).
// ---------------------------------------------------------------------------

typedef __attribute__((ext_vector_type(16))) __bf16 bf16x16;
typedef __attribute__((ext_vector_type(8)))  float  v8f;

// Pack two fp32 into a packed bf16 pair (RNE), lo in [15:0], hi in [31:16].
// RNE bias adds, then a single v_perm_b32 picks the two high halves.
__device__ __forceinline__ unsigned int pack_bf16(float lo, float hi) {
  unsigned int u0 = __float_as_uint(lo);
  unsigned int u1 = __float_as_uint(hi);
  u0 += 0x7FFFu + ((u0 >> 16) & 1u);
  u1 += 0x7FFFu + ((u1 >> 16) & 1u);
#if __has_builtin(__builtin_amdgcn_perm)
  return __builtin_amdgcn_perm(u1, u0, 0x07060302u);  // {u1.b3,u1.b2,u0.b3,u0.b2}
#else
  return (u0 >> 16) | (u1 & 0xFFFF0000u);
#endif
}

#define TBM 64
#define TBN 128
#define TBK 64

// Generic batched-strided GEMM: C[bz] = act(A[bz%aMod or bz] @ B[bz] + bias)
// Call sites guarantee M%64==0, N%128==0, K%64==0.
__global__ __launch_bounds__(256)
void gemm_bf16_wmma(const float* __restrict__ A, const float* __restrict__ Bm,
                    const float* __restrict__ bias, float* __restrict__ C,
                    int M, int N, int K, int lda, int ldb, int ldc,
                    long long strideA, long long strideB, long long strideC,
                    int aBatchMod, int doRelu)
{
  __shared__ __align__(16) unsigned short As[TBM][TBK + 2];   // [m][k] bf16 bits
  __shared__ __align__(16) unsigned short Bt[TBN][TBK + 2];   // [n][k] (B transposed)

  const int bz = blockIdx.z;
  const float* Ab = A  + (long long)(aBatchMod ? (bz % aBatchMod) : bz) * strideA;
  const float* Bb = Bm + (long long)bz * strideB;
  float*       Cb = C  + (long long)bz * strideC;

  const int m0 = blockIdx.y * TBM;
  const int n0 = blockIdx.x * TBN;
  const int tid  = threadIdx.x;
  const int w    = tid >> 5, lane = tid & 31;
  const int wm   = w & 1,  wn = w >> 1;       // 2 (M) x 4 (N) waves, 32x32 each
  const int g    = lane >> 4, lm = lane & 15; // lane half / lane-in-half

  v8f acc00 = {}, acc01 = {}, acc10 = {}, acc11 = {};

  for (int k0 = 0; k0 < K; k0 += TBK) {
    // ---- Stage A tile (64 x 64 fp32): float4 loads, packed-pair LDS stores.
    #pragma unroll
    for (int i = tid; i < (TBM * TBK) / 4; i += 256) {
      const int r = i >> 4, c4 = (i & 15) << 2;
      const float4 v = *(const float4*)&Ab[(long long)(m0 + r) * lda + (k0 + c4)];
      *(unsigned int*)&As[r][c4]     = pack_bf16(v.x, v.y);
      *(unsigned int*)&As[r][c4 + 2] = pack_bf16(v.z, v.w);
    }
    // ---- Stage B tile (64 x 128 fp32) transposed into [n][k]: each thread
    // takes a k-pair x n-quad, packs along k in registers, 32-bit LDS stores.
    #pragma unroll
    for (int i = tid; i < (TBK / 2) * (TBN / 4); i += 256) {
      const int kr = (i >> 5) << 1, c4 = (i & 31) << 2;
      const float* bp = &Bb[(long long)(k0 + kr) * ldb + (n0 + c4)];
      const float4 v0 = *(const float4*)bp;
      const float4 v1 = *(const float4*)(bp + ldb);
      *(unsigned int*)&Bt[c4 + 0][kr] = pack_bf16(v0.x, v1.x);
      *(unsigned int*)&Bt[c4 + 1][kr] = pack_bf16(v0.y, v1.y);
      *(unsigned int*)&Bt[c4 + 2][kr] = pack_bf16(v0.z, v1.z);
      *(unsigned int*)&Bt[c4 + 3][kr] = pack_bf16(v0.w, v1.w);
    }
    // ---- Prefetch next K-tile (global_prefetch_b8).
    if (k0 + TBK < K) {
      __builtin_prefetch(&Ab[(long long)(m0 + (tid & 63)) * lda + (k0 + TBK)], 0, 0);
      __builtin_prefetch(&Bb[(long long)(k0 + TBK + (tid >> 2)) * ldb + n0], 0, 0);
    }
    __syncthreads();

    // ---- 2 k-sub-steps x 4 WMMAs: 8 v_wmma per staging round.
    #pragma unroll
    for (int ks = 0; ks < TBK; ks += 32) {
      union Frag { bf16x16 v; unsigned int p[8]; } fa0, fa1, fb0, fb1;
      // Element j of lane (g,lm) holds K = j + 8g + (j>=8 ? 8 : 0);
      // pairs are K-contiguous -> 32-bit LDS pair loads (ds_load_2addr_b32).
      #pragma unroll
      for (int jp = 0; jp < 8; ++jp) {
        const int j  = jp << 1;
        const int kk = ks + j + 8 * g + ((j >= 8) ? 8 : 0);
        fa0.p[jp] = *(const unsigned int*)&As[wm * 32 + lm][kk];
        fa1.p[jp] = *(const unsigned int*)&As[wm * 32 + 16 + lm][kk];
        fb0.p[jp] = *(const unsigned int*)&Bt[wn * 32 + lm][kk];
        fb1.p[jp] = *(const unsigned int*)&Bt[wn * 32 + 16 + lm][kk];
      }
      acc00 = __builtin_amdgcn_wmma_f32_16x16x32_bf16(false, fa0.v, false, fb0.v,
                                                      (short)0, acc00, false, false);
      acc01 = __builtin_amdgcn_wmma_f32_16x16x32_bf16(false, fa0.v, false, fb1.v,
                                                      (short)0, acc01, false, false);
      acc10 = __builtin_amdgcn_wmma_f32_16x16x32_bf16(false, fa1.v, false, fb0.v,
                                                      (short)0, acc10, false, false);
      acc11 = __builtin_amdgcn_wmma_f32_16x16x32_bf16(false, fa1.v, false, fb1.v,
                                                      (short)0, acc11, false, false);
    }
    __syncthreads();
  }

  // ---- Epilogue. C/D layout: VGPR r -> M = r + 8g, N = lane&15.
  const int nC0 = n0 + wn * 32 + lm;
  const int nC1 = nC0 + 16;
  const float bv0 = bias ? bias[nC0] : 0.0f;
  const float bv1 = bias ? bias[nC1] : 0.0f;
  const int mBase = m0 + wm * 32 + g * 8;
  #pragma unroll
  for (int r = 0; r < 8; ++r) {
    float v00 = acc00[r] + bv0;
    float v01 = acc01[r] + bv1;
    float v10 = acc10[r] + bv0;
    float v11 = acc11[r] + bv1;
    if (doRelu) {
      v00 = fmaxf(v00, 0.0f); v01 = fmaxf(v01, 0.0f);
      v10 = fmaxf(v10, 0.0f); v11 = fmaxf(v11, 0.0f);
    }
    const long long row0 = mBase + r;
    const long long row1 = mBase + 16 + r;
    Cb[row0 * ldc + nC0] = v00;
    Cb[row0 * ldc + nC1] = v01;
    Cb[row1 * ldc + nC0] = v10;
    Cb[row1 * ldc + nC1] = v11;
  }
}

// ---------------------------------------------------------------------------
// Softmax attention over time, heads/nodes as batch. hd = 16.
// grid = (N, B, H), block = 64 (thread = query row l).
// ---------------------------------------------------------------------------
__global__ __launch_bounds__(64)
void attn_kernel(const float* __restrict__ Q, const float* __restrict__ Kv,
                 const float* __restrict__ Vv, float* __restrict__ O,
                 int Lq, int Lk)
{
  constexpr int Nn = 512, Dn = 128;
  const int n = blockIdx.x, b = blockIdx.y, h = blockIdx.z;
  __shared__ float Ks[96][17];
  __shared__ float Vs[96][17];
  const int tid = threadIdx.x;

  for (int i = tid; i < Lk * 16; i += 64) {
    int m = i >> 4, kk = i & 15;
    long long idx = ((((long long)b * Lk + m) * Nn) + n) * Dn + h * 16 + kk;
    Ks[m][kk] = Kv[idx];
    Vs[m][kk] = Vv[idx];
  }
  __syncthreads();
  if (tid >= Lq) return;

  const int l = tid;
  const long long qidx = ((((long long)b * Lq + l) * Nn) + n) * Dn + h * 16;
  float q[16];
  #pragma unroll
  for (int kk = 0; kk < 16; ++kk) q[kk] = Q[qidx + kk];

  float s[96];
  float mx = -1e30f;
  for (int m = 0; m < Lk; ++m) {
    float d = 0.0f;
    #pragma unroll
    for (int kk = 0; kk < 16; ++kk) d += q[kk] * Ks[m][kk];
    d *= 0.25f;                               // 1/sqrt(16)
    s[m] = d;
    mx = fmaxf(mx, d);
  }
  float sum = 0.0f;
  for (int m = 0; m < Lk; ++m) { float e = __expf(s[m] - mx); s[m] = e; sum += e; }
  const float inv = 1.0f / sum;

  float o[16];
  #pragma unroll
  for (int kk = 0; kk < 16; ++kk) o[kk] = 0.0f;
  for (int m = 0; m < Lk; ++m) {
    const float a = s[m] * inv;
    #pragma unroll
    for (int kk = 0; kk < 16; ++kk) o[kk] += a * Vs[m][kk];
  }
  #pragma unroll
  for (int kk = 0; kk < 16; ++kk) O[qidx + kk] = o[kk];
}

// ---------------------------------------------------------------------------
// Fused residual + LayerNorm over rows of 128. One wave32 per row (4 f32/lane),
// reductions via __shfl_xor. grid = rows/8, block = 256.
// ---------------------------------------------------------------------------
__global__ __launch_bounds__(256)
void ln_kernel(const float* __restrict__ A, const float* __restrict__ Bres,
               const float* __restrict__ gam, const float* __restrict__ bet,
               float* __restrict__ O, long long rows)
{
  const long long row = (long long)blockIdx.x * 8 + (threadIdx.x >> 5);
  if (row >= rows) return;
  const int lane = threadIdx.x & 31;

  float4 x = ((const float4*)(A + row * 128))[lane];
  if (Bres) {
    float4 y = ((const float4*)(Bres + row * 128))[lane];
    x.x += y.x; x.y += y.y; x.z += y.z; x.w += y.w;
  }
  float s  = x.x + x.y + x.z + x.w;
  float sq = x.x * x.x + x.y * x.y + x.z * x.z + x.w * x.w;
  #pragma unroll
  for (int off = 16; off > 0; off >>= 1) {
    s  += __shfl_xor(s,  off, 32);
    sq += __shfl_xor(sq, off, 32);
  }
  const float mean = s * (1.0f / 128.0f);
  const float var  = sq * (1.0f / 128.0f) - mean * mean;
  const float inv  = rsqrtf(var + 1e-5f);

  const int c = lane * 4;
  float4 o;
  o.x = (x.x - mean) * inv * gam[c + 0] + bet[c + 0];
  o.y = (x.y - mean) * inv * gam[c + 1] + bet[c + 1];
  o.z = (x.z - mean) * inv * gam[c + 2] + bet[c + 2];
  o.w = (x.w - mean) * inv * gam[c + 3] + bet[c + 3];
  ((float4*)(O + row * 128))[lane] = o;
}

// ---------------------------------------------------------------------------
// Batched 2D transpose with optional bias added per source row:
//   dst[bz*dStride + c*R + r] = src[bz*sStride + r*C + c] (+ bias[(bz%mod)*bs + r])
// ---------------------------------------------------------------------------
__global__ __launch_bounds__(256)
void transpose_bias_kernel(const float* __restrict__ S, float* __restrict__ Dst,
                           int R, int C, long long sStride, long long dStride,
                           const float* __restrict__ bias, int biasMod, int biasStride)
{
  __shared__ float t[32][33];
  const int bz = blockIdx.y;
  const int tilesR = (R + 31) >> 5;
  const int tr = (blockIdx.x % tilesR) * 32;
  const int tc = (blockIdx.x / tilesR) * 32;
  const float* Sp = S + (long long)bz * sStride;
  float*       Dp = Dst + (long long)bz * dStride;
  const int tid = threadIdx.x;

  for (int i = tid; i < 1024; i += 256) {
    int r = i >> 5, c = i & 31;
    int gr = tr + r, gc = tc + c;
    t[r][c] = (gr < R && gc < C) ? Sp[(long long)gr * C + gc] : 0.0f;
  }
  __syncthreads();
  for (int i = tid; i < 1024; i += 256) {
    int c = i >> 5, r = i & 31;             // write along r: coalesced
    int gc = tc + c, gr = tr + r;
    if (gc < C && gr < R) {
      float v = t[r][c];
      if (bias) v += bias[(long long)(bz % biasMod) * biasStride + gr];
      Dp[(long long)gc * R + gr] = v;
    }
  }
}

// ---------------------------------------------------------------------------
// Host-side orchestration
// ---------------------------------------------------------------------------
extern "C" void kernel_launch(void* const* d_in, const int* in_sizes, int n_in,
                              void* d_out, int out_size, void* d_ws, size_t ws_size,
                              hipStream_t stream)
{
  const float* x       = (const float*)d_in[0];
  const float* memory_ = (const float*)d_in[1];
  const float* data    = (const float*)d_in[2];
  const float* support = (const float*)d_in[3];
  const float* Wq_s = (const float*)d_in[4];
  const float* Wk_s = (const float*)d_in[5];
  const float* Wv_s = (const float*)d_in[6];
  const float* Wo_s = (const float*)d_in[7];
  const float* Wq_c = (const float*)d_in[8];
  const float* Wk_c = (const float*)d_in[9];
  const float* Wv_c = (const float*)d_in[10];
  const float* Wo_c = (const float*)d_in[11];
  const float* W1 = (const float*)d_in[12];
  const float* b1 = (const float*)d_in[13];
  const float* W2 = (const float*)d_in[14];
  const float* b2 = (const float*)d_in[15];
  const float* W3 = (const float*)d_in[16];
  const float* b3 = (const float*)d_in[17];
  const float* gcn_W = (const float*)d_in[18];
  const float* gcn_b = (const float*)d_in[19];
  const float* ln1_g = (const float*)d_in[20];
  const float* ln1_b = (const float*)d_in[21];
  const float* ln2_g = (const float*)d_in[22];
  const float* ln2_b = (const float*)d_in[23];
  const float* ln3_g = (const float*)d_in[24];
  const float* ln3_b = (const float*)d_in[25];

  constexpr int Bn = 4, Ln = 48, LPn = 96, Nn = 512, Dn = 128, DFn = 512;
  constexpr long long SZ_LND  = (long long)Bn * Ln  * Nn * Dn;   // 12.58M f32
  constexpr long long SZ_LPND = (long long)Bn * LPn * Nn * Dn;   // 25.17M f32

  float* ws  = (float*)d_ws;
  float* Qb  = ws;
  float* Kb  = Qb + SZ_LND;
  float* Vb  = Kb + SZ_LPND;
  float* Tb  = Vb + SZ_LPND;
  float* Ub  = Tb + SZ_LND;
  float* Hb  = Ub + SZ_LND;
  float* XNb = Hb + SZ_LND;
  float* Yb  = XNb + SZ_LND;
  float* Y1b = Yb + SZ_LND;        // [B,L,N,DF] = 50.3M f32
  float* Adyn = Y1b;               // alias: Y1 dead before A is written
  float* HCb  = Kb;                // alias: K+V+T (62.9M) == Hc size, dead by GCN phase
  float* Gb   = Qb;                // alias: Q dead by GCN phase

  float* out0 = (float*)d_out;
  float* dts  = out0 + SZ_LND;

  const int Mx   = Bn * Ln  * Nn;  // 98304
  const int Mmem = Bn * LPn * Nn;  // 196608
  const int BLn  = Bn * Ln;        // 192 (b,l) planes

  auto gemm = [&](const float* A, const float* Bw, const float* bias, float* C,
                  int M, int N, int K, int lda, int ldb, int ldc,
                  long long sA, long long sB, long long sC, int batch,
                  int aMod, int relu) {
    dim3 grid((unsigned)(N / TBN), (unsigned)(M / TBM), (unsigned)batch);
    gemm_bf16_wmma<<<grid, dim3(256), 0, stream>>>(A, Bw, bias, C, M, N, K,
                                                   lda, ldb, ldc, sA, sB, sC,
                                                   aMod, relu);
  };

  // ---- self-attention -----------------------------------------------------
  gemm(x, Wq_s, nullptr, Qb, Mx, Dn, Dn, Dn, Dn, Dn, 0, 0, 0, 1, 0, 0);
  gemm(x, Wk_s, nullptr, Kb, Mx, Dn, Dn, Dn, Dn, Dn, 0, 0, 0, 1, 0, 0);
  gemm(x, Wv_s, nullptr, Vb, Mx, Dn, Dn, Dn, Dn, Dn, 0, 0, 0, 1, 0, 0);
  attn_kernel<<<dim3(Nn, Bn, 8), dim3(64), 0, stream>>>(Qb, Kb, Vb, Tb, Ln, Ln);
  gemm(Tb, Wo_s, nullptr, Ub, Mx, Dn, Dn, Dn, Dn, Dn, 0, 0, 0, 1, 0, 0);
  ln_kernel<<<dim3(Mx / 8), dim3(256), 0, stream>>>(x, Ub, ln1_g, ln1_b, Hb, (long long)Mx);

  // ---- cross-attention ----------------------------------------------------
  gemm(Hb,      Wq_c, nullptr, Qb, Mx,   Dn, Dn, Dn, Dn, Dn, 0, 0, 0, 1, 0, 0);
  gemm(memory_, Wk_c, nullptr, Kb, Mmem, Dn, Dn, Dn, Dn, Dn, 0, 0, 0, 1, 0, 0);
  gemm(memory_, Wv_c, nullptr, Vb, Mmem, Dn, Dn, Dn, Dn, Dn, 0, 0, 0, 1, 0, 0);
  attn_kernel<<<dim3(Nn, Bn, 8), dim3(64), 0, stream>>>(Qb, Kb, Vb, Tb, Ln, LPn);
  gemm(Tb, Wo_c, nullptr, Ub, Mx, Dn, Dn, Dn, Dn, Dn, 0, 0, 0, 1, 0, 0);
  ln_kernel<<<dim3(Mx / 8), dim3(256), 0, stream>>>(Hb, Ub, ln2_g, ln2_b, XNb, (long long)Mx);

  // ---- FFN + output 0 -----------------------------------------------------
  gemm(XNb, W1, b1, Y1b, Mx, DFn, Dn,  Dn,  DFn, DFn, 0, 0, 0, 1, 0, 1);  // relu
  gemm(Y1b, W2, b2, Yb,  Mx, Dn,  DFn, DFn, Dn,  Dn,  0, 0, 0, 1, 0, 0);
  ln_kernel<<<dim3(Mx / 8), dim3(256), 0, stream>>>(XNb, Yb, ln3_g, ln3_b, out0, (long long)Mx);

  // ---- dynamic adjacency --------------------------------------------------
  gemm(Yb, W3, b3, Adyn, Mx, Nn, Dn, Dn, Nn, Nn, 0, 0, 0, 1, 0, 0);

  // ---- diffusion GCN ------------------------------------------------------
  const long long HCstride = (long long)5 * Dn * Nn;   // 327680 per (b,l)
  const long long SL       = (long long)Dn * Nn;       // 65536 slice
  // g0 = data^T -> Hc slice 0
  transpose_bias_kernel<<<dim3((Nn / 32) * (Dn / 32), BLn), dim3(256), 0, stream>>>(
      data, HCb, Nn, Dn, (long long)Nn * Dn, HCstride, nullptr, 1, 0);
  // s1, s2 (static support), d1, d2 (dynamic adjacency)
  gemm(HCb,          support, nullptr, HCb + SL,     Dn, Nn, Nn, Nn, Nn, Nn,
       HCstride, 0,                        HCstride, BLn, 0, 0);
  gemm(HCb + SL,     support, nullptr, HCb + 2 * SL, Dn, Nn, Nn, Nn, Nn, Nn,
       HCstride, 0,                        HCstride, BLn, 0, 0);
  gemm(HCb,          Adyn,    nullptr, HCb + 3 * SL, Dn, Nn, Nn, Nn, Nn, Nn,
       HCstride, (long long)Nn * Nn,       HCstride, BLn, 0, 0);
  gemm(HCb + 3 * SL, Adyn,    nullptr, HCb + 4 * SL, Dn, Nn, Nn, Nn, Nn, Nn,
       HCstride, (long long)Nn * Nn,       HCstride, BLn, 0, 0);
  // gout[b,l] = gcn_W[l] @ Hc[b,l]  (M=128, K=640, N=512; A batched by l)
  gemm(gcn_W, HCb, nullptr, Gb, Dn, Nn, 5 * Dn, 5 * Dn, Nn, Nn,
       (long long)Dn * 5 * Dn, HCstride, SL, BLn, Ln, 0);
  // dts = gout^T + gcn_b[l]
  transpose_bias_kernel<<<dim3((Dn / 32) * (Nn / 32), BLn), dim3(256), 0, stream>>>(
      Gb, dts, Dn, Nn, SL, SL, gcn_b, Ln, Dn);
}
